// MultiHeadAttentionBlock_22308060135775
// MI455X (gfx1250) — matrix operand order repaired and checked by reading, not compiled, not run on hardware
//
#include <hip/hip_runtime.h>

// ---------------------------------------------------------------------------
// MI455X (gfx1250, wave32) fused multi-head attention block.
//   q/k/v [4,2048,512] f32 -> QKV proj (f16 WMMA, BM=128) -> flash attention
//   (f16 WMMA, TDM K-tile staging, online softmax, /H scale + mask==0 -> -1e9)
//   -> out proj (f32).
// ---------------------------------------------------------------------------

typedef __attribute__((ext_vector_type(16))) _Float16 v16h;
typedef __attribute__((ext_vector_type(8)))  _Float16 v8h;
typedef __attribute__((ext_vector_type(8)))  float    v8f;
typedef __attribute__((ext_vector_type(4)))  unsigned int u32x4;
typedef __attribute__((ext_vector_type(8)))  int i32x8;
typedef __attribute__((ext_vector_type(4)))  int i32x4;

constexpr int BATCH = 4;
constexpr int SEQ   = 2048;
constexpr int DIM   = 512;
constexpr int NH    = 8;
constexpr int HDK   = 64;   // DIM / NH

#if defined(__gfx1250__) && __has_builtin(__builtin_amdgcn_tensor_load_to_lds) && \
    __has_builtin(__builtin_amdgcn_s_wait_tensorcnt)
#define USE_TDM 1
#else
#define USE_TDM 0
#endif

// A fragment, 16x32 f16, row-major tile in LDS (ld in halfs, multiple of 8).
// wave32: row = lane&15; VGPR0-3 hold K = half*8+0..7, VGPR4-7 K = 16+half*8+0..7.
__device__ __forceinline__ v16h frag_a(const _Float16* base, int ld, int lane) {
  const int row = lane & 15, half = lane >> 4;
  const _Float16* p = base + row * ld + half * 8;
  v8h lo = *(const v8h*)(p);
  v8h hi = *(const v8h*)(p + 16);
  v16h a;
#pragma unroll
  for (int i = 0; i < 8; ++i) { a[i] = lo[i]; a[8 + i] = hi[i]; }
  return a;
}

// B fragment, 32x16 f16, staged COLUMN-MAJOR by N: tile[n][k].
// wave32: col = lane&15; elements 0-7 hold K = half*8+0..7, 8-15 K = 16+half*8+0..7.
__device__ __forceinline__ v16h frag_b(const _Float16* base, int ld, int lane) {
  const int col = lane & 15, half = lane >> 4;
  const _Float16* p = base + col * ld + half * 8;
  v8h lo = *(const v8h*)(p);
  v8h hi = *(const v8h*)(p + 16);
  v16h b;
#pragma unroll
  for (int i = 0; i < 8; ++i) { b[i] = lo[i]; b[8 + i] = hi[i]; }
  return b;
}

__device__ __forceinline__ v8f wmma_f16(v16h a, v16h b, v8f c) {
  return __builtin_amdgcn_wmma_f32_16x16x32_f16(false, a, false, b,
                                                (short)0, c, false, false);
}

#if USE_TDM
// Issue a TDM load of a 64(rows)x64(halfs) f16 row-major tile (memory row
// stride 64 halfs) into LDS with a padded row stride of 72 halfs:
// pad_interval = 32 DWORDs (one 128B row), pad_amount = 4 DWORDs (16B).
__device__ __forceinline__ void tdm_load_tile_64x64_f16(const _Float16* gsrc,
                                                        void* lds_dst) {
  const unsigned long long ga = (unsigned long long)gsrc;
  const unsigned lds = (unsigned)(unsigned long long)lds_dst;
  u32x4 g0;
  g0[0] = 1u;                                   // count=1, user descriptor
  g0[1] = lds;                                  // lds_addr (bytes)
  g0[2] = (unsigned)ga;                         // global_addr[31:0]
  g0[3] = (unsigned)((ga >> 32) & 0x01FFFFFFull) | (2u << 30);  // addr[56:32]|type=2
  i32x8 g1;
  g1[0] = (int)((1u << 16)                      // data_size = 2 bytes
              | (1u << 20)                      // pad_enable
              | (4u << 22)                      // pad_interval: 32 DWORDs
              | (3u << 25));                    // pad_amount: 4 DWORDs
  g1[1] = (int)(64u << 16);                     // tensor_dim0 = 64 (bits 79:48)
  g1[2] = (int)((unsigned)SEQ << 16);           // tensor_dim1 = SEQ (bits 111:80)
  g1[3] = (int)(64u << 16);                     // tile_dim0 = 64 (bits 127:112)
  g1[4] = 64;                                   // tile_dim1 = 64 (bits 143:128)
  g1[5] = 64;                                   // tensor_dim0_stride = 64
  g1[6] = 0;
  g1[7] = 0;
  i32x4 z4 = {};
  i32x8 z8 = {};
  __builtin_amdgcn_tensor_load_to_lds(g0, g1, z4, z4, z8, 0);
}
#endif

// ---------------------------------------------------------------------------
// K0: fp32 -> f16 weight conversion (all four weight matrices).
// ---------------------------------------------------------------------------
__global__ __launch_bounds__(256) void convert_w_kernel(
    const float* __restrict__ Wq, const float* __restrict__ Wk,
    const float* __restrict__ Wv, const float* __restrict__ Wo,
    _Float16* __restrict__ q16, _Float16* __restrict__ k16,
    _Float16* __restrict__ v16, _Float16* __restrict__ o16) {
  size_t i = (size_t)blockIdx.x * blockDim.x + threadIdx.x;
  if (i < (size_t)DIM * DIM) {
    q16[i] = (_Float16)Wq[i];
    k16[i] = (_Float16)Wk[i];
    v16[i] = (_Float16)Wv[i];
    o16[i] = (_Float16)Wo[i];
  }
}

// ---------------------------------------------------------------------------
// K1: projection out = (X @ W^T + b), written head-major [B,H,S,HDK] f16.
// BM=128, BN=64, BK=32; 256 threads = 8 waves, wave w owns rows w*16..w*16+15
// across all 64 N columns (4 accumulators, 4 WMMAs per K-step).
// ---------------------------------------------------------------------------
__global__ __launch_bounds__(256) void qkv_proj_kernel(
    const float* __restrict__ X, const _Float16* __restrict__ W16,
    const float* __restrict__ bias, _Float16* __restrict__ outh) {
  __shared__ __align__(16) _Float16 As[128][40];  // [m][k], pad 32->40
  __shared__ __align__(16) _Float16 Bs[64][40];   // [n][k] == W natural layout
  const int m0 = blockIdx.x * 128;
  const int h  = blockIdx.y;
  const int n0 = h * HDK;
  const int tid = threadIdx.x, lane = tid & 31, wave = tid >> 5;
  const int half = lane >> 4, col = lane & 15;

  // fixed vectorized staging slices
  const int ar = tid >> 1, ac = (tid & 1) * 16;   // A: 16 f32 -> 16 f16
  const int br = tid >> 2, bc = (tid & 3) * 8;    // B: 8 f16

  v8f acc[4] = {v8f{}, v8f{}, v8f{}, v8f{}};

  for (int k0 = 0; k0 < DIM; k0 += 32) {
    const float* xp = &X[(size_t)(m0 + ar) * DIM + k0 + ac];
    float4 f0 = *(const float4*)(xp + 0);
    float4 f1 = *(const float4*)(xp + 4);
    float4 f2 = *(const float4*)(xp + 8);
    float4 f3 = *(const float4*)(xp + 12);
    v8h h0, h1;
    h0[0] = (_Float16)f0.x; h0[1] = (_Float16)f0.y;
    h0[2] = (_Float16)f0.z; h0[3] = (_Float16)f0.w;
    h0[4] = (_Float16)f1.x; h0[5] = (_Float16)f1.y;
    h0[6] = (_Float16)f1.z; h0[7] = (_Float16)f1.w;
    h1[0] = (_Float16)f2.x; h1[1] = (_Float16)f2.y;
    h1[2] = (_Float16)f2.z; h1[3] = (_Float16)f2.w;
    h1[4] = (_Float16)f3.x; h1[5] = (_Float16)f3.y;
    h1[6] = (_Float16)f3.z; h1[7] = (_Float16)f3.w;
    *(v8h*)&As[ar][ac]     = h0;
    *(v8h*)&As[ar][ac + 8] = h1;
    *(v8h*)&Bs[br][bc] = *(const v8h*)&W16[(size_t)(n0 + br) * DIM + k0 + bc];
    __syncthreads();
    v16h a = frag_a(&As[wave * 16][0], 40, lane);
#pragma unroll
    for (int t = 0; t < 4; ++t)
      acc[t] = wmma_f16(a, frag_b(&Bs[t * 16][0], 40, lane), acc[t]);
    __syncthreads();
  }

#pragma unroll
  for (int r = 0; r < 8; ++r) {
    const int m = m0 + wave * 16 + r + 8 * half;  // C/D: M = r + 8*half
    const int bb = m / SEQ, ss = m % SEQ;
    const size_t base = (((size_t)bb * NH + h) * SEQ + ss) * HDK;
#pragma unroll
    for (int t = 0; t < 4; ++t)
      outh[base + t * 16 + col] = (_Float16)(acc[t][r] + bias[n0 + t * 16 + col]);
  }
}

// ---------------------------------------------------------------------------
// K2: flash attention per (b, h, 64-query tile). 128 threads = 4 waves,
// each wave owns 16 query rows across all 64 keys of the current key tile.
// K tile staged by the Tensor Data Mover (TDM) when available.
// ---------------------------------------------------------------------------
__global__ __launch_bounds__(128) void attn_kernel(
    const _Float16* __restrict__ Qh, const _Float16* __restrict__ Kh,
    const _Float16* __restrict__ Vh, const int* __restrict__ mask,
    _Float16* __restrict__ ctx) {
  __shared__ __align__(16) _Float16 Qs[64][72];  // [q][dk]
  __shared__ __align__(16) _Float16 Ks[64][72];  // [key][dk] (B col-major)
  __shared__ __align__(16) _Float16 Vt[64][72];  // [dk][key] (B col-major)
  __shared__ __align__(16) _Float16 Ps[64][72];  // [q][key] f16 probs
  __shared__ float Ss[64][66];
  __shared__ float mrow[64], lrow[64], arow[64];

  const int q0 = blockIdx.x * 64;
  const int h  = blockIdx.y;
  const int b  = blockIdx.z;
  const size_t hb = ((size_t)b * NH + h) * (size_t)SEQ * HDK;

  const int tid = threadIdx.x, lane = tid & 31, wave = tid >> 5;
  const int half = lane >> 4, col = lane & 15;

  // fixed vectorized staging slices: 32 halfs per thread
  const int sr = tid >> 1, sc = (tid & 1) * 32;

  {
    const _Float16* qp = &Qh[hb + (size_t)(q0 + sr) * HDK + sc];
#pragma unroll
    for (int j = 0; j < 4; ++j)
      *(v8h*)&Qs[sr][sc + 8 * j] = *(const v8h*)(qp + 8 * j);
  }
  if (tid < 64) { mrow[tid] = -3.0e38f; lrow[tid] = 0.0f; }

  v8f o0 = {}, o1 = {}, o2 = {}, o3 = {};
  const float scale = 1.0f / (float)NH;          // faithful quirk: /num_heads

  for (int kt = 0; kt < SEQ / 64; ++kt) {
    const int kg0 = kt * 64;
    __syncthreads();                             // prior tiles fully consumed

#if USE_TDM
    if (wave == 0)                               // one TDM op stages all of Ks
      tdm_load_tile_64x64_f16(&Kh[hb + (size_t)kg0 * HDK], &Ks[0][0]);
#else
    {
      const _Float16* kp = &Kh[hb + (size_t)(kg0 + sr) * HDK + sc];
#pragma unroll
      for (int j = 0; j < 4; ++j)
        *(v8h*)&Ks[sr][sc + 8 * j] = *(const v8h*)(kp + 8 * j);
    }
#endif
    {  // V transposed on stage: vector global loads, scalar LDS stores
      const _Float16* vp = &Vh[hb + (size_t)(kg0 + sr) * HDK + sc];
#pragma unroll
      for (int j = 0; j < 4; ++j) {
        v8h vv = *(const v8h*)(vp + 8 * j);
#pragma unroll
        for (int e = 0; e < 8; ++e) Vt[sc + 8 * j + e][sr] = vv[e];
      }
    }
    if (kt + 1 < SEQ / 64)                       // warm next V tile
      __builtin_prefetch(&Vh[hb + (size_t)(kg0 + 64 + sr) * HDK], 0, 1);
#if USE_TDM
    if (wave == 0) __builtin_amdgcn_s_wait_tensorcnt(0);
#endif
    __syncthreads();

    // --- S = Q K^T (K-dim = dk = 64 -> two WMMA k-steps x 4 key tiles) ---
    v8f s0 = {}, s1 = {}, s2 = {}, s3 = {};
#pragma unroll
    for (int kk = 0; kk < HDK; kk += 32) {
      v16h a = frag_a(&Qs[wave * 16][kk], 72, lane);
      s0 = wmma_f16(a, frag_b(&Ks[ 0][kk], 72, lane), s0);
      s1 = wmma_f16(a, frag_b(&Ks[16][kk], 72, lane), s1);
      s2 = wmma_f16(a, frag_b(&Ks[32][kk], 72, lane), s2);
      s3 = wmma_f16(a, frag_b(&Ks[48][kk], 72, lane), s3);
    }

    // --- scale + mask, spill to LDS for the row-wise softmax ---
#pragma unroll
    for (int r = 0; r < 8; ++r) {
      const int ml = wave * 16 + r + 8 * half;
      const size_t mb = ((size_t)b * SEQ + (q0 + ml)) * (size_t)SEQ + kg0;
      Ss[ml][ 0 + col] = mask[mb +  0 + col] ? s0[r] * scale : -1.0e9f;
      Ss[ml][16 + col] = mask[mb + 16 + col] ? s1[r] * scale : -1.0e9f;
      Ss[ml][32 + col] = mask[mb + 32 + col] ? s2[r] * scale : -1.0e9f;
      Ss[ml][48 + col] = mask[mb + 48 + col] ? s3[r] * scale : -1.0e9f;
    }
    __syncthreads();

    // --- online softmax: one thread per query row ---
    if (tid < 64) {
      const int r = tid;
      float mo = mrow[r], mx = mo;
#pragma unroll 8
      for (int c2 = 0; c2 < 64; ++c2) mx = fmaxf(mx, Ss[r][c2]);
      const float al = __expf(mo - mx);
      float l = lrow[r] * al;
#pragma unroll 8
      for (int c2 = 0; c2 < 64; ++c2) {
        float p = __expf(Ss[r][c2] - mx);
        Ps[r][c2] = (_Float16)p;
        l += p;
      }
      mrow[r] = mx; lrow[r] = l; arow[r] = al;
    }
    __syncthreads();

    // --- rescale running O, then O += P V ---
#pragma unroll
    for (int r = 0; r < 8; ++r) {
      const float al = arow[wave * 16 + r + 8 * half];
      o0[r] *= al; o1[r] *= al; o2[r] *= al; o3[r] *= al;
    }
#pragma unroll
    for (int kk = 0; kk < 64; kk += 32) {        // K-dim = key index
      v16h a = frag_a(&Ps[wave * 16][kk], 72, lane);
      o0 = wmma_f16(a, frag_b(&Vt[ 0][kk], 72, lane), o0);
      o1 = wmma_f16(a, frag_b(&Vt[16][kk], 72, lane), o1);
      o2 = wmma_f16(a, frag_b(&Vt[32][kk], 72, lane), o2);
      o3 = wmma_f16(a, frag_b(&Vt[48][kk], 72, lane), o3);
    }
  }
  __syncthreads();

  // --- epilogue: O / l, write ctx in [B,S,D] f16 for the out-proj ---
#pragma unroll
  for (int r = 0; r < 8; ++r) {
    const int ml = wave * 16 + r + 8 * half;
    const float li = 1.0f / lrow[ml];
    const size_t ob = ((size_t)b * SEQ + (q0 + ml)) * DIM + h * HDK;
    ctx[ob +  0 + col] = (_Float16)(o0[r] * li);
    ctx[ob + 16 + col] = (_Float16)(o1[r] * li);
    ctx[ob + 32 + col] = (_Float16)(o2[r] * li);
    ctx[ob + 48 + col] = (_Float16)(o3[r] * li);
  }
}

// ---------------------------------------------------------------------------
// K3: output projection out = ctx @ Wo^T + bo (f16 WMMA, fp32 result), BM=128.
// ---------------------------------------------------------------------------
__global__ __launch_bounds__(256) void out_proj_kernel(
    const _Float16* __restrict__ Xh, const _Float16* __restrict__ W16,
    const float* __restrict__ bias, float* __restrict__ out) {
  __shared__ __align__(16) _Float16 As[128][40];
  __shared__ __align__(16) _Float16 Bs[64][40];
  const int m0 = blockIdx.x * 128;
  const int n0 = blockIdx.y * 64;
  const int tid = threadIdx.x, lane = tid & 31, wave = tid >> 5;
  const int half = lane >> 4, col = lane & 15;

  const int ar = tid >> 1, ac = (tid & 1) * 16;  // A: 16 f16 per thread
  const int br = tid >> 2, bc = (tid & 3) * 8;   // B: 8 f16 per thread

  v8f acc[4] = {v8f{}, v8f{}, v8f{}, v8f{}};

  for (int k0 = 0; k0 < DIM; k0 += 32) {
    const _Float16* xp = &Xh[(size_t)(m0 + ar) * DIM + k0 + ac];
    *(v8h*)&As[ar][ac]     = *(const v8h*)(xp + 0);
    *(v8h*)&As[ar][ac + 8] = *(const v8h*)(xp + 8);
    *(v8h*)&Bs[br][bc] = *(const v8h*)&W16[(size_t)(n0 + br) * DIM + k0 + bc];
    __syncthreads();
    v16h a = frag_a(&As[wave * 16][0], 40, lane);
#pragma unroll
    for (int t = 0; t < 4; ++t)
      acc[t] = wmma_f16(a, frag_b(&Bs[t * 16][0], 40, lane), acc[t]);
    __syncthreads();
  }

#pragma unroll
  for (int r = 0; r < 8; ++r) {
    const int m = m0 + wave * 16 + r + 8 * half;
#pragma unroll
    for (int t = 0; t < 4; ++t)
      out[(size_t)m * DIM + n0 + t * 16 + col] = acc[t][r] + bias[n0 + t * 16 + col];
  }
}

// ---------------------------------------------------------------------------
// Host-side launcher. Workspace (f16): Wq16|Wk16|Wv16|Wo16 (4*512*512) then
// qh|kh|vh|ctx (4 * B*S*D) ~ 36 MB.
// ---------------------------------------------------------------------------
extern "C" void kernel_launch(void* const* d_in, const int* in_sizes, int n_in,
                              void* d_out, int out_size, void* d_ws, size_t ws_size,
                              hipStream_t stream) {
  (void)in_sizes; (void)n_in; (void)out_size; (void)ws_size;
  const float* q    = (const float*)d_in[0];
  const float* k    = (const float*)d_in[1];
  const float* v    = (const float*)d_in[2];
  const int*   mask = (const int*)  d_in[3];
  const float* Wq   = (const float*)d_in[4];
  const float* bq   = (const float*)d_in[5];
  const float* Wk   = (const float*)d_in[6];
  const float* bk   = (const float*)d_in[7];
  const float* Wv   = (const float*)d_in[8];
  const float* bv   = (const float*)d_in[9];
  const float* Wo   = (const float*)d_in[10];
  const float* bo   = (const float*)d_in[11];
  float* out = (float*)d_out;

  const size_t DD  = (size_t)DIM * DIM;
  const size_t BSD = (size_t)BATCH * SEQ * DIM;
  _Float16* ws   = (_Float16*)d_ws;
  _Float16* Wq16 = ws;
  _Float16* Wk16 = Wq16 + DD;
  _Float16* Wv16 = Wk16 + DD;
  _Float16* Wo16 = Wv16 + DD;
  _Float16* qh   = Wo16 + DD;
  _Float16* kh   = qh + BSD;
  _Float16* vh   = kh + BSD;
  _Float16* ctx  = vh + BSD;

  convert_w_kernel<<<dim3((unsigned)((DD + 255) / 256)), 256, 0, stream>>>(
      Wq, Wk, Wv, Wo, Wq16, Wk16, Wv16, Wo16);

  dim3 gProj(BATCH * SEQ / 128, NH);            // 64 x 8 blocks
  qkv_proj_kernel<<<gProj, 256, 0, stream>>>(q, Wq16, bq, qh);
  qkv_proj_kernel<<<gProj, 256, 0, stream>>>(k, Wk16, bk, kh);
  qkv_proj_kernel<<<gProj, 256, 0, stream>>>(v, Wv16, bv, vh);

  attn_kernel<<<dim3(SEQ / 64, NH, BATCH), 128, 0, stream>>>(qh, kh, vh, mask, ctx);

  out_proj_kernel<<<dim3(BATCH * SEQ / 128, DIM / 64), 256, 0, stream>>>(ctx, Wo16, bo, out);
}
